// Model2_27625229648357
// MI455X (gfx1250) — compile-verified, matrix operand order checked by reading, not run on hardware
//
#include <hip/hip_runtime.h>
#include <hip/hip_bf16.h>
#include <math.h>

typedef __attribute__((ext_vector_type(16))) _Float16     v16h;
typedef __attribute__((ext_vector_type(8)))  _Float16     v8h;
typedef __attribute__((ext_vector_type(8)))  float        v8f;
typedef __attribute__((ext_vector_type(4)))  unsigned int u32x4;
typedef __attribute__((ext_vector_type(8)))  int          i32x8;
typedef __attribute__((ext_vector_type(4)))  int          i32x4;

#define WS_OFF_B   (100u * 1024u * 1024u)               /* buffer B region   */
#define WS_OFF_W   (WS_OFF_B + 80u * 1024u * 1024u)     /* packed weights    */

// ---------------------------------------------------------------------------
// Tensor Data Mover: load a 2D f16 tile [d1 rows][d0 contiguous elems] from
// global (row stride = stride0 elems) into LDS, rows packed densely.
// D# packing per CDNA5 ISA ch.8; all fields wave-uniform -> SGPR groups.
// ---------------------------------------------------------------------------
__device__ __forceinline__ void tdm_load_tile_f16(void* lds_dst, const _Float16* gsrc,
                                                  int d0, int d1, int stride0) {
    unsigned long long ga = (unsigned long long)(uintptr_t)gsrc;
    unsigned int lds_off  = (unsigned int)(uintptr_t)lds_dst;   // low 32 = LDS offset

    u32x4 g0;
    g0[0] = 1u;                                            // count=1, user desc
    g0[1] = lds_off;                                       // lds_addr
    g0[2] = (unsigned int)(ga & 0xFFFFFFFFu);              // global_addr[31:0]
    g0[3] = (unsigned int)((ga >> 32) & 0x01FFFFFFu)       // global_addr[56:32]
            | (2u << 30);                                  // type=2 ("image")

    i32x8 g1;
    g1[0] = (1 << 16);                                     // data_size=1 -> 2 bytes
    g1[1] = (int)((d0 & 0xFFFF) << 16);                    // tensor_dim0[15:0]
    g1[2] = (int)(((unsigned)d0 >> 16)                     // tensor_dim0[31:16]
            | ((unsigned)(d1 & 0xFFFF) << 16));            // tensor_dim1[15:0]
    g1[3] = (int)((((unsigned)d1 >> 16) & 0xFFFFu)         // tensor_dim1[31:16]
            | ((unsigned)(d0 & 0xFFFF) << 16));            // tile_dim0 = d0
    g1[4] = (int)(d1 & 0xFFFF);                            // tile_dim1 (tile_dim2=0)
    g1[5] = stride0;                                       // tensor_dim0_stride[31:0]
    g1[6] = 0;                                             // stride0[47:32] | s1 lo
    g1[7] = 0;                                             // tensor_dim1_stride hi

    i32x4 z4 = {0, 0, 0, 0};
    i32x8 z8 = {0, 0, 0, 0, 0, 0, 0, 0};
    __builtin_amdgcn_tensor_load_to_lds(g0, g1, z4, z4, z8, 0);
}

__device__ __forceinline__ v16h cat8(v8h lo, v8h hi) {
    return __builtin_shufflevector(lo, hi, 0, 1, 2, 3, 4, 5, 6, 7,
                                   8, 9, 10, 11, 12, 13, 14, 15);
}

// ---------------------------------------------------------------------------
// f32 -> f16 activation conversion (grid-stride)
// ---------------------------------------------------------------------------
__global__ __launch_bounds__(256) void cvt_f32_f16(const float* __restrict__ src,
                                                   _Float16* __restrict__ dst, long n) {
    long i = blockIdx.x * (long)blockDim.x + threadIdx.x;
    long step = (long)gridDim.x * blockDim.x;
    for (; i < n; i += step) dst[i] = (_Float16)src[i];
}

// ---------------------------------------------------------------------------
// Repack weights N-major: dst[n*Kpad + k] (zero padded).
// kMajor=1: src[k*N+n] (FC weights, transpose); kMajor=0: src[n*K+k] (conv).
// ---------------------------------------------------------------------------
__global__ __launch_bounds__(256) void prep_w(const float* __restrict__ src,
                                              _Float16* __restrict__ dst,
                                              int K, int N, int Kpad, int Npad, int kMajor) {
    int i = blockIdx.x * blockDim.x + threadIdx.x;
    int tot = Kpad * Npad;
    int step = gridDim.x * blockDim.x;
    for (; i < tot; i += step) {
        int n = i / Kpad, k = i - n * Kpad;
        float v = 0.f;
        if (k < K && n < N) v = kMajor ? src[k * N + n] : src[n * K + k];
        dst[i] = (_Float16)v;
    }
}

// ---------------------------------------------------------------------------
// Implicit-GEMM 4D conv + bias + ReLU via v_wmma_f32_16x16x32_f16.
// One wave = one 16(M)x16(N) output tile.  Grids divide exactly, so EXEC is
// all-ones at every WMMA / barrier.  Weights are staged to LDS by the Tensor
// Data Mover, double-buffered: chunk i+1 is in flight (s_wait_tensorcnt 1)
// while chunk i is consumed via ds_load_b128 fragments.
// ---------------------------------------------------------------------------
#define CONV_CHUNK 256

template <int KSZ>
__global__ __launch_bounds__(256)
void conv_wmma(const _Float16* __restrict__ in, const _Float16* __restrict__ wT,
               const float* __restrict__ bias, _Float16* __restrict__ out,
               int Ci, int Din, int Co, int Dout, int Kpad, int Mtot) {
    constexpr int K4 = KSZ * KSZ * KSZ * KSZ;
    constexpr int K3 = KSZ * KSZ * KSZ;
    constexpr int K2 = KSZ * KSZ;

    __shared__ __align__(16) _Float16 sB[2][CONV_CHUNK * 16];   // 2 x 8 KB

    const int wave  = threadIdx.x >> 5;
    const int lane  = threadIdx.x & 31;
    const int mtile = blockIdx.x * (blockDim.x >> 5) + wave;
    const int m0    = mtile * 16;

    const int laneM = lane & 15;
    const int hi8   = (lane >> 4) & 1;
    const int kHalf = hi8 * 8;

    const int spatial = Dout * Dout * Dout * Dout;
    const int D2 = Din * Din, D3 = D2 * Din, D4 = D3 * Din;

    int m  = m0 + laneM;                                   // always < Mtot (exact grid)
    int b  = m / spatial, s = m - b * spatial;
    int o4 = s % Dout; int t = s / Dout;
    int o3 = t % Dout; t /= Dout;
    int o2 = t % Dout; int o1 = t / Dout;
    const int baseA = b * Ci * D4 + o1 * D3 + o2 * D2 + o3 * Din + o4;

    const int nch = (Kpad + CONV_CHUNK - 1) / CONV_CHUNK;
    if (wave == 0) {
        int r0 = Kpad < CONV_CHUNK ? Kpad : CONV_CHUNK;
        tdm_load_tile_f16(&sB[0][0], wT, r0, 16, Kpad);    // preload chunk 0
    }

    v8f c = {};
    for (int ic = 0; ic < nch; ++ic) {
        const int kc   = ic * CONV_CHUNK;
        int rows = Kpad - kc; if (rows > CONV_CHUNK) rows = CONV_CHUNK;  // mult of 32

        if (wave == 0) {
            if (ic + 1 < nch) {                            // launch next chunk
                int kc2 = kc + CONV_CHUNK;
                int r2  = Kpad - kc2; if (r2 > CONV_CHUNK) r2 = CONV_CHUNK;
                tdm_load_tile_f16(&sB[(ic + 1) & 1][0], wT + kc2, r2, 16, Kpad);
                __builtin_amdgcn_s_wait_tensorcnt(1);      // chunk ic landed
            } else {
                __builtin_amdgcn_s_wait_tensorcnt(0);
            }
        }
        __syncthreads();                                   // publish chunk ic

        const _Float16* sb = &sB[ic & 1][0];
        for (int k0 = kc; k0 < kc + rows; k0 += 32) {
            if (k0 + 32 < Kpad)
                __builtin_prefetch(&in[baseA], 0, 1);      // global_prefetch_b8
            v16h a;
#pragma unroll
            for (int e = 0; e < 16; ++e) {
                // 16-bit A-fragment layout: K = e (+8 for e>=8) (+8 for hi lanes)
                int kl = e + ((e & 8) ? 8 : 0) + kHalf;
                int kg = k0 + kl;
                int ci = kg / K4; int r = kg - ci * K4;    // shifts when KSZ==4
                int k1 = r / K3;  r -= k1 * K3;
                int k2 = r / K2;  r -= k2 * K2;
                int k3 = r / KSZ; int kk4 = r - k3 * KSZ;
                a[e] = (ci < Ci)
                    ? in[baseA + ci * D4 + k1 * D3 + k2 * D2 + k3 * Din + kk4]
                    : (_Float16)0.f;
            }
            // B fragment: two contiguous 8xf16 runs per lane -> 2x ds_load_b128
            const _Float16* p = sb + laneM * rows + (k0 - kc) + kHalf;
            v16h bm = cat8(*(const v8h*)p, *(const v8h*)(p + 16));
            c = __builtin_amdgcn_wmma_f32_16x16x32_f16(false, a, false, bm,
                                                       (short)0, c, false, false);
        }
        __syncthreads();                                   // readers done with buf
    }

    const float bn = (laneM < Co) ? bias[laneM] : 0.f;
#pragma unroll
    for (int r = 0; r < 8; ++r) {
        int   mo = m0 + r + hi8 * 8;                       // C layout: VGPR r -> row
        float v  = c[r] + bn;
        v = v > 0.f ? v : 0.f;
        if (laneM < Co) {
            int b2 = mo / spatial, s2 = mo - b2 * spatial;
            out[((long)b2 * Co + laneM) * spatial + s2] = (_Float16)v;
        }
    }
}

// ---------------------------------------------------------------------------
// Dense GEMM (FC1): out[M][Npad] = relu(A[M][Kpad] * W^T + bias),
// W stored N-major [Npad][Kpad].  TDM double-buffered chunks of 128 K x 112 N.
// Grid covers exactly (M/16)*(Npad/16) tiles.
// ---------------------------------------------------------------------------
#define FC_CHUNK 128

__global__ __launch_bounds__(256)
void gemm_wmma(const _Float16* __restrict__ A, const _Float16* __restrict__ wT,
               const float* __restrict__ bias, _Float16* __restrict__ out,
               int M, int Kpad, int Npad, int N, int relu) {
    __shared__ __align__(16) _Float16 sB[2][FC_CHUNK * 112];    // 2 x 28 KB

    const int wave = threadIdx.x >> 5;
    const int lane = threadIdx.x & 31;
    const int tile = blockIdx.x * (blockDim.x >> 5) + wave;
    const int ntiles = Npad >> 4;
    const int tm = tile / ntiles, tn = tile - tm * ntiles;

    const int laneM = lane & 15;
    const int hi8   = (lane >> 4) & 1;
    const int kHalf = hi8 * 8;
    int m = tm * 16 + laneM;
    int n = tn * 16 + laneM;

    const int nch = (Kpad + FC_CHUNK - 1) / FC_CHUNK;
    if (wave == 0) {
        int r0 = Kpad < FC_CHUNK ? Kpad : FC_CHUNK;
        tdm_load_tile_f16(&sB[0][0], wT, r0, Npad, Kpad);
    }

    v8f c = {};
    for (int ic = 0; ic < nch; ++ic) {
        const int kc = ic * FC_CHUNK;
        int rows = Kpad - kc; if (rows > FC_CHUNK) rows = FC_CHUNK;

        if (wave == 0) {
            if (ic + 1 < nch) {
                int kc2 = kc + FC_CHUNK;
                int r2  = Kpad - kc2; if (r2 > FC_CHUNK) r2 = FC_CHUNK;
                tdm_load_tile_f16(&sB[(ic + 1) & 1][0], wT + kc2, r2, Npad, Kpad);
                __builtin_amdgcn_s_wait_tensorcnt(1);
            } else {
                __builtin_amdgcn_s_wait_tensorcnt(0);
            }
        }
        __syncthreads();

        const _Float16* sb = &sB[ic & 1][0];
        for (int k0 = kc; k0 < kc + rows; k0 += 32) {
            v16h a;
#pragma unroll
            for (int e = 0; e < 16; ++e) {
                int kg = k0 + e + ((e & 8) ? 8 : 0) + kHalf;
                a[e] = A[(long)m * Kpad + kg];
            }
            const _Float16* p = sb + n * rows + (k0 - kc) + kHalf;
            v16h bm = cat8(*(const v8h*)p, *(const v8h*)(p + 16));
            c = __builtin_amdgcn_wmma_f32_16x16x32_f16(false, a, false, bm,
                                                       (short)0, c, false, false);
        }
        __syncthreads();
    }

    const float bn = (n < N) ? bias[n] : 0.f;
#pragma unroll
    for (int r = 0; r < 8; ++r) {
        int   mo = tm * 16 + r + hi8 * 8;
        float v  = c[r] + bn;
        if (relu) v = v > 0.f ? v : 0.f;
        out[(long)mo * Npad + n] = (_Float16)v;
    }
}

// ---------------------------------------------------------------------------
// FC2 (100 -> 1) + sigmoid; tiny, scalar per output row.
// ---------------------------------------------------------------------------
__global__ __launch_bounds__(256)
void fc2_sigmoid(const _Float16* __restrict__ h, const float* __restrict__ w,
                 const float* __restrict__ b, float* __restrict__ out,
                 int M, int K, int stride) {
    int i = blockIdx.x * blockDim.x + threadIdx.x;
    if (i >= M) return;
    float acc = b[0];
    for (int k = 0; k < K; ++k) acc += (float)h[i * stride + k] * w[k];
    out[i] = 1.f / (1.f + expf(-acc));
}

// ---------------------------------------------------------------------------
extern "C" void kernel_launch(void* const* d_in, const int* in_sizes, int n_in,
                              void* d_out, int out_size, void* d_ws, size_t ws_size,
                              hipStream_t stream) {
    const float* x     = (const float*)d_in[0];
    const float* w1    = (const float*)d_in[1];  const float* b1 = (const float*)d_in[2];
    const float* w2    = (const float*)d_in[3];  const float* b2 = (const float*)d_in[4];
    const float* w3    = (const float*)d_in[5];  const float* b3 = (const float*)d_in[6];
    const float* w4    = (const float*)d_in[7];  const float* b4 = (const float*)d_in[8];
    const float* w5    = (const float*)d_in[9];  const float* b5 = (const float*)d_in[10];
    const float* fc1w  = (const float*)d_in[11]; const float* fc1b = (const float*)d_in[12];
    const float* fc2w  = (const float*)d_in[13]; const float* fc2b = (const float*)d_in[14];
    float* out = (float*)d_out;

    char* ws = (char*)d_ws;
    _Float16* bufA = (_Float16*)(ws);                 // acts 0,2,4 + fc1 out
    _Float16* bufB = (_Float16*)(ws + WS_OFF_B);      // acts 1,3,5
    char* wsw = ws + WS_OFF_W;
    _Float16* wB1  = (_Float16*)(wsw);                // 16 x  256
    _Float16* wB2  = (_Float16*)(wsw + 8192);         // 16 x  768
    _Float16* wB3  = (_Float16*)(wsw + 32768);        // 16 x 2304
    _Float16* wB4  = (_Float16*)(wsw + 106496);       // 16 x 3072
    _Float16* wB5  = (_Float16*)(wsw + 204800);       // 16 x 1216
    _Float16* wF1  = (_Float16*)(wsw + 243712);       // 112 x 3840
    _Float16* wF2  = (_Float16*)(wsw + 1103872);      // 16 x 128

    const int B = 256;

    // --- convert input activations to f16 ---
    long nx = (long)B * 1 * 18 * 18 * 18 * 18;        // 26,873,856
    cvt_f32_f16<<<8192, 256, 0, stream>>>(x, bufA, nx);

    // --- repack weights N-major [Npad][Kpad] (conv: direct; fc: transpose) ---
    prep_w<<<16,   256, 0, stream>>>(w1,   wB1,  256, 3,  256, 16, 0);
    prep_w<<<48,   256, 0, stream>>>(w2,   wB2,  768, 9,  768, 16, 0);
    prep_w<<<144,  256, 0, stream>>>(w3,   wB3, 2304, 12, 2304, 16, 0);
    prep_w<<<192,  256, 0, stream>>>(w4,   wB4, 3072, 15, 3072, 16, 0);
    prep_w<<<76,   256, 0, stream>>>(w5,   wB5, 1215, 15, 1216, 16, 0);
    prep_w<<<1680, 256, 0, stream>>>(fc1w, wF1, 3840, 100, 3840, 112, 1);
    prep_w<<<8,    256, 0, stream>>>(fc2w, wF2,  100, 1,  128, 16, 1);

    // --- conv stack: 8 waves/block, 128 output rows/block (grids divide exactly) ---
    conv_wmma<4><<<101250, 256, 0, stream>>>(bufA, wB1, b1, bufB, 1, 18, 3, 15, 256, 12960000);
    conv_wmma<4><<<41472, 256, 0, stream>>>(bufB, wB2, b2, bufA, 3, 15, 9, 12, 768, 5308416);
    conv_wmma<4><<<13122, 256, 0, stream>>>(bufA, wB3, b3, bufB, 9, 12, 12, 9, 2304, 1679616);
    conv_wmma<4><<<2592, 256, 0, stream>>>(bufB, wB4, b4, bufA, 12, 9, 15, 6, 3072, 331776);
    conv_wmma<3><<<512, 256, 0, stream>>>(bufA, wB5, b5, bufB, 15, 6, 15, 4, 1216, 65536);

    // --- FC1: [256 x 3840] * [3840 x 112pad] + relu; 16*7=112 tiles, 8 waves/block ---
    gemm_wmma<<<14, 256, 0, stream>>>(bufB, wF1, fc1b, bufA, 256, 3840, 112, 100, 1);

    // --- FC2 + sigmoid ---
    fc2_sigmoid<<<1, 256, 0, stream>>>(bufA, fc2w, fc2b, out, 256, 100, 112);
}
